// Cross_modal_attention_32341103738984
// MI455X (gfx1250) — matrix operand order repaired and checked by reading, not compiled, not run on hardware
//
#include <hip/hip_runtime.h>
#include <hip/hip_bf16.h>

typedef __attribute__((ext_vector_type(16))) __bf16 v16bf;
typedef __attribute__((ext_vector_type(8)))  float  v8f;

union BF16x16 { v16bf v; unsigned int u[8]; };

__device__ __forceinline__ unsigned short f2bf(float f) {
    unsigned u = __float_as_uint(f);
    u += 0x7FFFu + ((u >> 16) & 1u);          // round-to-nearest-even
    return (unsigned short)(u >> 16);
}
__device__ __forceinline__ float bf2f(unsigned short h) {
    return __uint_as_float(((unsigned)h) << 16);
}
__device__ __forceinline__ unsigned pk2(float a, float b) {
    return (unsigned)f2bf(a) | ((unsigned)f2bf(b) << 16);
}

// ---------------------------------------------------------------------------
// Core: 32(M) x 512(N) x 512(K) bf16 GEMM tile.
// A staged in LDS [32][520] bf16 (padded: row stride 1040B -> conflict-free),
// B = weight [e][d] row-major bf16 in global (L2-resident).
// 8 waves, wave w owns N-strip [64w, 64w+64): 2 m-tiles x 4 n-tiles = 8 acc.
// ---------------------------------------------------------------------------
__device__ __forceinline__ void run_gemm_512(
    const unsigned short (*sA)[520], const unsigned short* __restrict__ Wb,
    int lane, int n0w, v8f acc[2][4])
{
    const int half = lane >> 4;
    const int mr   = lane & 15;
    for (int k0 = 0; k0 < 512; k0 += 32) {
        BF16x16 aF[2];
#pragma unroll
        for (int m = 0; m < 2; ++m) {
            const uint4* p = reinterpret_cast<const uint4*>(&sA[m * 16 + mr][k0 + 8 * half]);
            uint4 x = p[0], y = p[2];
            aF[m].u[0] = x.x; aF[m].u[1] = x.y; aF[m].u[2] = x.z; aF[m].u[3] = x.w;
            aF[m].u[4] = y.x; aF[m].u[5] = y.y; aF[m].u[6] = y.z; aF[m].u[7] = y.w;
        }
#pragma unroll
        for (int t = 0; t < 4; ++t) {
            const int e = n0w + t * 16 + mr;
            const uint4* q = reinterpret_cast<const uint4*>(Wb + (size_t)e * 512 + k0 + 8 * half);
            uint4 x = q[0], y = q[2];
            BF16x16 bF;
            bF.u[0] = x.x; bF.u[1] = x.y; bF.u[2] = x.z; bF.u[3] = x.w;
            bF.u[4] = y.x; bF.u[5] = y.y; bF.u[6] = y.z; bF.u[7] = y.w;
#pragma unroll
            for (int m = 0; m < 2; ++m)
                acc[m][t] = __builtin_amdgcn_wmma_f32_16x16x32_bf16(
                    false, aF[m].v, false, bF.v, (short)0, acc[m][t], false, false);
        }
    }
}

// ---------------------------------------------------------------------------
// Kernel 0: fp32 -> bf16 weight conversion
// ---------------------------------------------------------------------------
__global__ __launch_bounds__(256) void cvt_bf16_kernel(
    const float* __restrict__ in, unsigned short* __restrict__ out, int n)
{
    for (int i = blockIdx.x * blockDim.x + threadIdx.x; i < n; i += gridDim.x * blockDim.x)
        out[i] = f2bf(in[i]);
}

// ---------------------------------------------------------------------------
// Kernel 1: O = l2norm_row(X @ W^T + bias); optionally Adot = (O @ w_g)*scale
// ---------------------------------------------------------------------------
__global__ __launch_bounds__(256) void qk_gemm_kernel(
    const float* __restrict__ X, const unsigned short* __restrict__ Wb,
    const float* __restrict__ bias, const float* __restrict__ wg,
    unsigned short* __restrict__ Obf, float* __restrict__ Adot, int withDot)
{
    __shared__ __align__(16) unsigned short sA[32][520];
    __shared__ float part[8][32];
    __shared__ float invRow[32];

    const int tid = threadIdx.x;
    const long m0 = (long)blockIdx.x * 32;

    // stage 32x512 fp32 rows -> bf16 LDS (b128 in, b128 out)
    for (int base = tid * 8; base < 32 * 512; base += 256 * 8) {
        const int row = base >> 9, d = base & 511;
        const float4* ap = reinterpret_cast<const float4*>(X + (m0 + row) * 512 + d);
        float4 u = ap[0], v = ap[1];
        uint4 pk;
        pk.x = pk2(u.x, u.y); pk.y = pk2(u.z, u.w);
        pk.z = pk2(v.x, v.y); pk.w = pk2(v.z, v.w);
        *reinterpret_cast<uint4*>(&sA[row][d]) = pk;
    }
    __syncthreads();

    const int w = tid >> 5, lane = tid & 31, n0w = w * 64;
    const int half = lane >> 4, mr = lane & 15;

    v8f acc[2][4];
    const v8f z = {0.f, 0.f, 0.f, 0.f, 0.f, 0.f, 0.f, 0.f};
#pragma unroll
    for (int m = 0; m < 2; ++m)
#pragma unroll
        for (int t = 0; t < 4; ++t) acc[m][t] = z;

    run_gemm_512(sA, Wb, lane, n0w, acc);

    // bias + per-row sum of squares (this wave's 64 columns)
    float s[2][8] = {};
#pragma unroll
    for (int m = 0; m < 2; ++m)
#pragma unroll
        for (int t = 0; t < 4; ++t) {
            const float be = bias[n0w + t * 16 + mr];
#pragma unroll
            for (int r = 0; r < 8; ++r) {
                acc[m][t][r] += be;
                s[m][r] += acc[m][t][r] * acc[m][t][r];
            }
        }
#pragma unroll
    for (int m = 0; m < 2; ++m)
#pragma unroll
        for (int r = 0; r < 8; ++r) {
            float v = s[m][r];
            v += __shfl_xor(v, 1, 32);
            v += __shfl_xor(v, 2, 32);
            v += __shfl_xor(v, 4, 32);
            v += __shfl_xor(v, 8, 32);
            s[m][r] = v;
        }
    if (mr == 0) {
#pragma unroll
        for (int m = 0; m < 2; ++m)
#pragma unroll
            for (int r = 0; r < 8; ++r) part[w][m * 16 + 8 * half + r] = s[m][r];
    }
    __syncthreads();
    if (tid < 32) {
        float t2 = 0.f;
#pragma unroll
        for (int w2 = 0; w2 < 8; ++w2) t2 += part[w2][tid];
        invRow[tid] = 1.f / fmaxf(sqrtf(t2), 1e-12f);
    }
    __syncthreads();

    // normalize, store bf16, accumulate gating dot
    float dp[2][8] = {};
#pragma unroll
    for (int m = 0; m < 2; ++m)
#pragma unroll
        for (int t = 0; t < 4; ++t) {
            const int e = n0w + t * 16 + mr;
            const float wgv = withDot ? wg[e] : 0.f;
#pragma unroll
            for (int r = 0; r < 8; ++r) {
                const int row = m * 16 + 8 * half + r;
                const float v = acc[m][t][r] * invRow[row];
                Obf[(m0 + row) * 512 + e] = f2bf(v);
                dp[m][r] += v * wgv;
            }
        }
    if (withDot) {
#pragma unroll
        for (int m = 0; m < 2; ++m)
#pragma unroll
            for (int r = 0; r < 8; ++r) {
                float v = dp[m][r];
                v += __shfl_xor(v, 1, 32);
                v += __shfl_xor(v, 2, 32);
                v += __shfl_xor(v, 4, 32);
                v += __shfl_xor(v, 8, 32);
                dp[m][r] = v;
            }
        if (mr == 0) {
#pragma unroll
            for (int m = 0; m < 2; ++m)
#pragma unroll
                for (int r = 0; r < 8; ++r) part[w][m * 16 + 8 * half + r] = dp[m][r];
        }
        __syncthreads();
        if (tid < 32) {
            float t2 = 0.f;
#pragma unroll
            for (int w2 = 0; w2 < 8; ++w2) t2 += part[w2][tid];
            Adot[m0 + tid] = t2 * 0.0441941738241592f; // D^-0.5
        }
    }
}

// ---------------------------------------------------------------------------
// Kernel 2a: invA[b] = 1 / max(||A[b,:]||_2, eps)          (8 blocks)
// ---------------------------------------------------------------------------
__global__ __launch_bounds__(256) void anorm_kernel(
    const float* __restrict__ Abuf, float* __restrict__ invA)
{
    __shared__ float red[256];
    const int b = blockIdx.x, tid = threadIdx.x;
    float s = 0.f;
    for (int i = tid; i < 4096; i += 256) {
        const float v = Abuf[b * 4096 + i];
        s += v * v;
    }
    red[tid] = s;
    __syncthreads();
    for (int off = 128; off > 0; off >>= 1) {
        if (tid < off) red[tid] += red[tid + off];
        __syncthreads();
    }
    if (tid == 0) invA[b] = 1.f / fmaxf(sqrtf(red[0]), 1e-12f);
}

// ---------------------------------------------------------------------------
// Kernel 2b: Gpart[b,c,d] = sum over 256-row chunk c of A[n]*q[n,d]
//            grid = B*16 blocks -> 128 blocks, fully parallel over N
// ---------------------------------------------------------------------------
__global__ __launch_bounds__(256) void pool_partial_kernel(
    const float* __restrict__ Abuf, const unsigned short* __restrict__ qbf,
    float* __restrict__ Gpart)
{
    __shared__ float Ah[256];
    const int blk = blockIdx.x;           // b*16 + chunk
    const int b = blk >> 4, chunk = blk & 15;
    const int tid = threadIdx.x;
    const long n0 = (long)b * 4096 + chunk * 256;

    Ah[tid] = Abuf[n0 + tid];
    __syncthreads();

    const unsigned short* qb = qbf + n0 * 512;
    float acc0 = 0.f, acc1 = 0.f;
    for (int n = 0; n < 256; ++n) {
        const float an = Ah[n];
        acc0 += an * bf2f(qb[(size_t)n * 512 + tid]);
        acc1 += an * bf2f(qb[(size_t)n * 512 + tid + 256]);
    }
    Gpart[(size_t)blk * 512 + tid]       = acc0;
    Gpart[(size_t)blk * 512 + tid + 256] = acc1;
}

// ---------------------------------------------------------------------------
// Kernel 2c: G[b,d] = invA[b] * sum_c Gpart[b,c,d]         (16 blocks)
// ---------------------------------------------------------------------------
__global__ __launch_bounds__(256) void pool_combine_kernel(
    const float* __restrict__ Gpart, const float* __restrict__ invA,
    float* __restrict__ G)
{
    const int idx = blockIdx.x * 256 + threadIdx.x;   // b*512 + d
    const int b = idx >> 9, d = idx & 511;
    float s = 0.f;
#pragma unroll
    for (int c = 0; c < 16; ++c)
        s += Gpart[((size_t)(b * 16 + c)) * 512 + d];
    G[idx] = s * invA[b];
}

// ---------------------------------------------------------------------------
// Kernel 3: t = (G*key) @ Wp^T + bp + query ;  out = t @ Wf^T + bf
// Two chained WMMA GEMMs, intermediate t lives only in LDS (bf16).
// ---------------------------------------------------------------------------
__global__ __launch_bounds__(256) void fused_out_kernel(
    const unsigned short* __restrict__ kbf, const unsigned short* __restrict__ qbf,
    const float* __restrict__ G, const unsigned short* __restrict__ Wpb,
    const float* __restrict__ bp, const unsigned short* __restrict__ Wfb,
    const float* __restrict__ bfv, float* __restrict__ out)
{
    __shared__ __align__(16) unsigned short sA[32][520];
    __shared__ __align__(16) unsigned short sT[32][520];

    const int tid = threadIdx.x;
    const long m0 = (long)blockIdx.x * 32;
    const float* Gr = G + (m0 >> 12) * 512;   // batch = m0 / 4096

    // stage gk = G (.) key  as bf16
    for (int base = tid * 8; base < 32 * 512; base += 256 * 8) {
        const int row = base >> 9, d = base & 511;
        const uint4 kk = *reinterpret_cast<const uint4*>(kbf + (m0 + row) * 512 + d);
        const float4 g0 = *reinterpret_cast<const float4*>(Gr + d);
        const float4 g1 = *reinterpret_cast<const float4*>(Gr + d + 4);
        uint4 pk;
        pk.x = pk2(bf2f(kk.x & 0xffff) * g0.x, bf2f(kk.x >> 16) * g0.y);
        pk.y = pk2(bf2f(kk.y & 0xffff) * g0.z, bf2f(kk.y >> 16) * g0.w);
        pk.z = pk2(bf2f(kk.z & 0xffff) * g1.x, bf2f(kk.z >> 16) * g1.y);
        pk.w = pk2(bf2f(kk.w & 0xffff) * g1.z, bf2f(kk.w >> 16) * g1.w);
        *reinterpret_cast<uint4*>(&sA[row][d]) = pk;
    }
    __syncthreads();

    const int w = tid >> 5, lane = tid & 31, n0w = w * 64;
    const int half = lane >> 4, mr = lane & 15;

    v8f acc[2][4];
    const v8f z = {0.f, 0.f, 0.f, 0.f, 0.f, 0.f, 0.f, 0.f};
#pragma unroll
    for (int m = 0; m < 2; ++m)
#pragma unroll
        for (int t = 0; t < 4; ++t) acc[m][t] = z;

    run_gemm_512(sA, Wpb, lane, n0w, acc);

    // epilogue 1: + bp + query residual -> bf16 into sT
#pragma unroll
    for (int m = 0; m < 2; ++m)
#pragma unroll
        for (int t = 0; t < 4; ++t) {
            const int e = n0w + t * 16 + mr;
            const float be = bp[e];
#pragma unroll
            for (int r = 0; r < 8; ++r) {
                const int row = m * 16 + 8 * half + r;
                const float v = acc[m][t][r] + be + bf2f(qbf[(m0 + row) * 512 + e]);
                sT[row][e] = f2bf(v);
            }
        }
    __syncthreads();

    v8f acc2[2][4];
#pragma unroll
    for (int m = 0; m < 2; ++m)
#pragma unroll
        for (int t = 0; t < 4; ++t) acc2[m][t] = z;

    run_gemm_512(sT, Wfb, lane, n0w, acc2);

#pragma unroll
    for (int m = 0; m < 2; ++m)
#pragma unroll
        for (int t = 0; t < 4; ++t) {
            const int e = n0w + t * 16 + mr;
            const float be = bfv[e];
#pragma unroll
            for (int r = 0; r < 8; ++r) {
                const int row = m * 16 + 8 * half + r;
                out[(m0 + row) * 512 + e] = acc2[m][t][r] + be;
            }
        }
}

// ---------------------------------------------------------------------------
extern "C" void kernel_launch(void* const* d_in, const int* in_sizes, int n_in,
                              void* d_out, int out_size, void* d_ws, size_t ws_size,
                              hipStream_t stream) {
    (void)n_in; (void)out_size; (void)ws_size;
    const float* a   = (const float*)d_in[0];
    const float* b   = (const float*)d_in[1];
    const float* Wq  = (const float*)d_in[2];
    const float* bq  = (const float*)d_in[3];
    const float* Wk  = (const float*)d_in[4];
    const float* bk  = (const float*)d_in[5];
    const float* wg  = (const float*)d_in[6];
    const float* Wp  = (const float*)d_in[7];
    const float* bp  = (const float*)d_in[8];
    const float* Wf  = (const float*)d_in[9];
    const float* bfv = (const float*)d_in[10];
    float* out = (float*)d_out;

    const int D = 512, Nn = 4096;
    const int Bb = in_sizes[0] / (Nn * D);
    const long M = (long)Bb * Nn;

    char* ws = (char*)d_ws;
    size_t off = 0;
    unsigned short* qbf = (unsigned short*)(ws + off); off += (size_t)M * D * 2;
    unsigned short* kbf = (unsigned short*)(ws + off); off += (size_t)M * D * 2;
    unsigned short* Wqb = (unsigned short*)(ws + off); off += (size_t)D * D * 2;
    unsigned short* Wkb = (unsigned short*)(ws + off); off += (size_t)D * D * 2;
    unsigned short* Wpb = (unsigned short*)(ws + off); off += (size_t)D * D * 2;
    unsigned short* Wfb = (unsigned short*)(ws + off); off += (size_t)D * D * 2;
    float* Abuf  = (float*)(ws + off); off += (size_t)M * 4;
    float* G     = (float*)(ws + off); off += (size_t)Bb * D * 4;
    float* Gpart = (float*)(ws + off); off += (size_t)Bb * 16 * D * 4;
    float* invA  = (float*)(ws + off);

    cvt_bf16_kernel<<<256, 256, 0, stream>>>(Wq, Wqb, D * D);
    cvt_bf16_kernel<<<256, 256, 0, stream>>>(Wk, Wkb, D * D);
    cvt_bf16_kernel<<<256, 256, 0, stream>>>(Wp, Wpb, D * D);
    cvt_bf16_kernel<<<256, 256, 0, stream>>>(Wf, Wfb, D * D);

    const int nTiles = (int)(M / 32);
    qk_gemm_kernel<<<nTiles, 256, 0, stream>>>(a, Wqb, bq, wg, qbf, Abuf, 1);
    qk_gemm_kernel<<<nTiles, 256, 0, stream>>>(b, Wkb, bk, wg, kbf, Abuf, 0);

    anorm_kernel<<<Bb, 256, 0, stream>>>(Abuf, invA);
    pool_partial_kernel<<<Bb * 16, 256, 0, stream>>>(Abuf, qbf, Gpart);
    pool_combine_kernel<<<Bb * 2, 256, 0, stream>>>(Gpart, invA, G);

    fused_out_kernel<<<nTiles, 256, 0, stream>>>(kbf, qbf, G, Wpb, bp, Wfb, bfv, out);
}
